// Attention_87033217286882
// MI455X (gfx1250) — matrix operand order repaired and checked by reading, not compiled, not run on hardware
//
#include <hip/hip_runtime.h>
#include <hip/hip_bf16.h>
#include <stdint.h>

// ---------------------------------------------------------------------------
// MI455X (gfx1250, wave32) fused attention layer.
// [QKV GEMM (bf16 WMMA) + scatter] -> [RoPE] -> [flash attention (bf16 WMMA
// for QK^T and PV, f32 online softmax)] -> [out-proj GEMM].
// Data movement: LDS staging with 16B ds/global ops; async GLOBAL->LDS copies
// (GLOBAL_LOAD_ASYNC_TO_LDS_B128, ASYNCcnt) where no dtype conversion is
// needed; global_prefetch for the next stage.
// All matmuls: v_wmma_f32_16x16x32_bf16.
// ---------------------------------------------------------------------------

typedef __bf16 bf16;
typedef __attribute__((ext_vector_type(16))) __bf16 v16bf;
typedef __attribute__((ext_vector_type(8)))  __bf16 v8bf;
typedef __attribute__((ext_vector_type(8)))  float  v8f;

// Copy 16 bytes global -> LDS per lane via the CDNA5 async-to-LDS path
// (no VGPR round trip; tracked by ASYNCcnt).  The LDS operand is the low
// 32 bits of the generic pointer (flat LDS address = aperture | offset,
// ISA 10.2).
__device__ __forceinline__ void copy16_g2l(const bf16* g, bf16* l) {
  const uint32_t lds_addr = (uint32_t)(uintptr_t)l;
  const uint64_t gaddr = (uint64_t)(uintptr_t)g;
  asm volatile("global_load_async_to_lds_b128 %0, %1, off"
               :
               : "v"(lds_addr), "v"(gaddr)
               : "memory");
}
__device__ __forceinline__ void async_fence() {
  asm volatile("s_wait_asynccnt 0" ::: "memory");
}

__device__ __forceinline__ v8f wmma_bf16f32(v16bf a, v16bf b, v8f c) {
  // 8-arg form: (neg_a, A, neg_b, B, c_mod, C, reuse_a, reuse_b)
  return __builtin_amdgcn_wmma_f32_16x16x32_bf16(false, a, false, b,
                                                 (short)0, c, false, false);
}

// A fragment (16x32, MxK), memory row-major [M][ldk] (ldk mult of 8).
// ISA layout: lanes 0-15 row M=lane, K 0..7 / 16..23; lanes 16-31 same rows,
// K 8..15 / 24..31.  -> two contiguous 16B loads per lane.
__device__ __forceinline__ v16bf load_frag_a(const bf16* base, int ldk) {
  const int lane = threadIdx.x & 31;
  const int m  = lane & 15;
  const int hi = lane >> 4;
  const bf16* p = base + (size_t)m * ldk;
  v8bf lo = *(const v8bf*)(p + hi * 8);
  v8bf hv = *(const v8bf*)(p + 16 + hi * 8);
  v16bf r;
#pragma unroll
  for (int e = 0; e < 8; ++e) { r[e] = lo[e]; r[e + 8] = hv[e]; }
  return r;
}

// B fragment (32x16, KxN), memory laid out [N][ldk] (contiguous K per col n).
// lanes 0-15: K=0..7 / 16..23, lanes 16-31: K=8..15 / 24..31.
__device__ __forceinline__ v16bf load_frag_b(const bf16* base, int ldk) {
  const int lane = threadIdx.x & 31;
  const int n  = lane & 15;
  const int hi = lane >> 4;
  const bf16* p = base + (size_t)n * ldk + hi * 8;
  v8bf lo = *(const v8bf*)(p);
  v8bf hv = *(const v8bf*)(p + 16);
  v16bf r;
#pragma unroll
  for (int e = 0; e < 8; ++e) { r[e] = lo[e]; r[e + 8] = hv[e]; }
  return r;
}

// Row reductions across the 16 lanes holding one C-layout row (half-wave).
__device__ __forceinline__ float half_max(float v) {
#pragma unroll
  for (int m = 1; m < 16; m <<= 1) v = fmaxf(v, __shfl_xor(v, m, 16));
  return v;
}
__device__ __forceinline__ float half_sum(float v) {
#pragma unroll
  for (int m = 1; m < 16; m <<= 1) v += __shfl_xor(v, m, 16);
  return v;
}

// ---------------------------------------------------------------------------
// Tiled GEMM: C[M,N] = A[M,K] * Bw[K,N] + bias.  128x128 WG tile, K-stage 64
// (2 WMMA K-steps per barrier).  8 waves as 4(M) x 2(N); wave = 32x64 tile.
// MODE 0: scatter bf16 into q/k/v [B,H,S,64].  MODE 1: f32 store to Cout.
// ---------------------------------------------------------------------------
template<bool A_BF16, int MODE>
__global__ __launch_bounds__(256) void gemm_bf16_wmma(
    const void* __restrict__ Ap, const float* __restrict__ Bw,
    const float* __restrict__ bias, float* __restrict__ Cout,
    bf16* __restrict__ qp, bf16* __restrict__ kp, bf16* __restrict__ vp,
    int M, int N, int K, int S)
{
  constexpr int LDT = 72;   // 128x64 tiles, padded rows (144B, 16B aligned)
  __shared__ bf16 smA[128 * LDT];
  __shared__ bf16 smB[128 * LDT];

  const int tid  = threadIdx.x;
  const int lane = tid & 31;
  const int wid  = tid >> 5;
  const int hi   = lane >> 4;
  const int cl   = lane & 15;
  const int m0 = blockIdx.x * 128;
  const int n0 = blockIdx.y * 128;
  const int wm = (wid & 3) * 32;
  const int wn = (wid >> 2) * 64;

  v8f acc[2][4];
#pragma unroll
  for (int a = 0; a < 2; ++a)
#pragma unroll
    for (int t = 0; t < 4; ++t) acc[a][t] = (v8f){0,0,0,0,0,0,0,0};

  const int ar = tid >> 1, ac = (tid & 1) * 32;   // A: 2 thr/row, 32 elem
  const int br = tid >> 2, bc = (tid & 3) * 32;   // B: 4 thr/row, 32 elem

  for (int k0 = 0; k0 < K; k0 += 64) {
    // Prefetch next stage's global lines (global_prefetch_b8).
    if (k0 + 64 < K) {
      if (A_BF16)
        __builtin_prefetch((const bf16*)Ap + (size_t)(m0 + ar) * K + k0 + 64 + ac, 0, 1);
      else
        __builtin_prefetch((const float*)Ap + (size_t)(m0 + ar) * K + k0 + 64 + ac, 0, 1);
      __builtin_prefetch(Bw + (size_t)(k0 + 64 + br) * N + n0 + bc, 0, 1);
    }
    // Stage A.
    if (A_BF16) {
      const bf16* ag = (const bf16*)Ap + (size_t)(m0 + ar) * K + k0 + ac;
      copy16_g2l(ag,      &smA[ar * LDT + ac]);
      copy16_g2l(ag + 16, &smA[ar * LDT + ac + 16]);
    } else {
      const float* ag = (const float*)Ap + (size_t)(m0 + ar) * K + k0 + ac;
#pragma unroll
      for (int j = 0; j < 32; ++j) smA[ar * LDT + ac + j] = (bf16)ag[j];
    }
    // Stage B: coalesced global read along N, transposed store -> LDS [n][k].
    const float* bg = Bw + (size_t)(k0 + br) * N + n0 + bc;
#pragma unroll
    for (int j = 0; j < 32; ++j) smB[(bc + j) * LDT + br] = (bf16)bg[j];
    if (A_BF16) async_fence();
    __syncthreads();

#pragma unroll
    for (int ks = 0; ks < 64; ks += 32) {
      v16bf a0 = load_frag_a(&smA[(wm + 0)  * LDT + ks], LDT);
      v16bf a1 = load_frag_a(&smA[(wm + 16) * LDT + ks], LDT);
#pragma unroll
      for (int t = 0; t < 4; ++t) {
        v16bf bt = load_frag_b(&smB[(wn + 16 * t) * LDT + ks], LDT);
        acc[0][t] = wmma_bf16f32(a0, bt, acc[0][t]);
        acc[1][t] = wmma_bf16f32(a1, bt, acc[1][t]);
      }
    }
    __syncthreads();
  }

  // Epilogue.  C layout: element i -> row (hi?8:0)+i of 16-tile, col = cl.
#pragma unroll
  for (int mt = 0; mt < 2; ++mt) {
#pragma unroll
    for (int t = 0; t < 4; ++t) {
      const int colN = n0 + wn + 16 * t + cl;
      const float bv = bias[colN];
#pragma unroll
      for (int i = 0; i < 8; ++i) {
        const int row = m0 + wm + 16 * mt + hi * 8 + i;
        const float val = acc[mt][t][i] + bv;
        if (MODE == 0) {
          const int which = colN >> 10;       // 0=q 1=k 2=v
          const int rem = colN & 1023;
          const int h = rem >> 6, d = rem & 63;
          const int bI = row / S, sI = row % S;
          bf16* dst = (which == 0) ? qp : ((which == 1) ? kp : vp);
          dst[((((size_t)bI * 16) + h) * (size_t)S + sI) * 64 + d] = (bf16)val;
        } else {
          Cout[(size_t)row * N + colN] = val;
        }
      }
    }
  }
}

// ---------------------------------------------------------------------------
// RoPE in place on q and k ([B,H,S,64]).  One thread per (tensor,b,h,s,i<32).
// ---------------------------------------------------------------------------
__global__ __launch_bounds__(256) void rope_kernel(bf16* __restrict__ qp,
                                                   bf16* __restrict__ kp,
                                                   int S)
{
  const size_t per = (size_t)4 * 16 * (size_t)S * 32;   // B*H*S*32
  size_t idx = (size_t)blockIdx.x * blockDim.x + threadIdx.x;
  bf16* arr = (idx < per) ? qp : kp;
  size_t r = (idx < per) ? idx : idx - per;
  const int i = (int)(r & 31);
  const size_t t = r >> 5;
  const int s = (int)(t % (size_t)S);
  const size_t bh = t / (size_t)S;
  // inv_freq = 10000^(-i/32) = exp(-ln(10000)/32 * i)
  const float ang = (float)s * __expf(-0.28782313663108517f * (float)i);
  float sn, c;
  __sincosf(ang, &sn, &c);
  const size_t base = (bh * (size_t)S + s) * 64;
  const float x1 = (float)arr[base + i];
  const float x2 = (float)arr[base + i + 32];
  arr[base + i]      = (bf16)(x1 * c - x2 * sn);
  arr[base + i + 32] = (bf16)(x2 * c + x1 * sn);
}

// ---------------------------------------------------------------------------
// Flash attention: WG = (b, h, 128 q rows); 8 waves own 16 q rows each.
// KV blocked by 64.  K staged natural [kv][d] (B operand for QK^T) via async
// GLOBAL->LDS; V staged transposed [d][kv] (B operand for PV).  P round-trips
// through a per-wave LDS patch (C-layout -> A-layout).
// ---------------------------------------------------------------------------
__global__ __launch_bounds__(256) void flash_attn(
    const bf16* __restrict__ qp, const bf16* __restrict__ kp,
    const bf16* __restrict__ vp, const unsigned char* __restrict__ kv_mask,
    bf16* __restrict__ attn, int S)
{
  constexpr int LDK = 72;   // padded (144B rows, 16B aligned)
  __shared__ bf16 smK [64 * LDK];
  __shared__ bf16 smVt[64 * LDK];
  __shared__ bf16 smP [8 * 16 * LDK];

  const int tid  = threadIdx.x;
  const int lane = tid & 31;
  const int wid  = tid >> 5;
  const int hi   = lane >> 4;
  const int cl   = lane & 15;
  const int qb = blockIdx.x * 128;
  const int h  = blockIdx.y;
  const int b  = blockIdx.z;

  const size_t head = ((size_t)b * 16 + h) * (size_t)S * 64;
  const bf16* qbase = qp + head;
  const bf16* kbase = kp + head;
  const bf16* vbase = vp + head;

  // Q fragments for this wave's 16 rows (d = 0..31 and 32..63), loaded once.
  const int qrow = qb + wid * 16;
  const v16bf qa0 = load_frag_a(qbase + (size_t)qrow * 64, 64);
  const v16bf qa1 = load_frag_a(qbase + (size_t)qrow * 64 + 32, 64);

  v8f out[4];
#pragma unroll
  for (int t = 0; t < 4; ++t) out[t] = (v8f){0,0,0,0,0,0,0,0};
  float mprev[8], lsum[8];
#pragma unroll
  for (int i = 0; i < 8; ++i) { mprev[i] = -1e30f; lsum[i] = 0.0f; }
  const float scale = 0.125f;   // 1/sqrt(64)

  const int ldr = tid >> 2, ldc = (tid & 3) * 16;   // KV loader: 4 thr/row

  for (int kv0 = 0; kv0 < S; kv0 += 64) {
    // Prefetch next KV block.
    if (kv0 + 64 < S) {
      __builtin_prefetch(kbase + (size_t)(kv0 + 64 + ldr) * 64 + ldc, 0, 1);
      __builtin_prefetch(vbase + (size_t)(kv0 + 64 + ldr) * 64 + ldc, 0, 1);
    }
    // Stage K natural (async GLOBAL->LDS), V transposed.
    {
      const bf16* kg = kbase + (size_t)(kv0 + ldr) * 64 + ldc;
      copy16_g2l(kg,     &smK[ldr * LDK + ldc]);
      copy16_g2l(kg + 8, &smK[ldr * LDK + ldc + 8]);
      const bf16* vg = vbase + (size_t)(kv0 + ldr) * 64 + ldc;
#pragma unroll
      for (int j = 0; j < 16; ++j) smVt[(ldc + j) * LDK + ldr] = vg[j];
    }
    async_fence();
    __syncthreads();

    // scores (16 q x 64 kv): 4 N-tiles x 2 K-steps of WMMA.
    v8f sc[4];
#pragma unroll
    for (int t = 0; t < 4; ++t) {
      v8f z = (v8f){0,0,0,0,0,0,0,0};
      sc[t] = wmma_bf16f32(qa0, load_frag_b(&smK[(16 * t) * LDK], LDK), z);
      sc[t] = wmma_bf16f32(qa1, load_frag_b(&smK[(16 * t) * LDK + 32], LDK), sc[t]);
    }

    // scale + mask (col = kv index lives on lane cl, same for all 8 rows)
#pragma unroll
    for (int t = 0; t < 4; ++t) {
      const bool mv = kv_mask[(size_t)b * S + kv0 + 16 * t + cl] != 0;
#pragma unroll
      for (int i = 0; i < 8; ++i)
        sc[t][i] = mv ? sc[t][i] * scale : -1e30f;
    }

    // online softmax
    float mnew[8], corr[8], rs[8];
#pragma unroll
    for (int i = 0; i < 8; ++i) {
      float r = fmaxf(fmaxf(sc[0][i], sc[1][i]), fmaxf(sc[2][i], sc[3][i]));
      r = half_max(r);
      mnew[i] = fmaxf(mprev[i], r);
      corr[i] = __expf(mprev[i] - mnew[i]);
      rs[i] = 0.0f;
    }
#pragma unroll
    for (int t = 0; t < 4; ++t)
#pragma unroll
      for (int i = 0; i < 8; ++i) {
        const float p = __expf(sc[t][i] - mnew[i]);
        sc[t][i] = p;
        rs[i] += p;
      }
#pragma unroll
    for (int i = 0; i < 8; ++i) {
      rs[i] = half_sum(rs[i]);
      lsum[i] = lsum[i] * corr[i] + rs[i];
      mprev[i] = mnew[i];
    }
#pragma unroll
    for (int t = 0; t < 4; ++t)
#pragma unroll
      for (int i = 0; i < 8; ++i) out[t][i] *= corr[i];

    // P: C-layout -> per-wave LDS patch [16][64] -> A fragments.
    bf16* pbase = &smP[wid * 16 * LDK];
#pragma unroll
    for (int t = 0; t < 4; ++t)
#pragma unroll
      for (int i = 0; i < 8; ++i)
        pbase[(hi * 8 + i) * LDK + 16 * t + cl] = (bf16)sc[t][i];

    // out += P (16 x 64kv) * V (64kv x 64d)
#pragma unroll
    for (int s2 = 0; s2 < 2; ++s2) {
      v16bf pa = load_frag_a(pbase + s2 * 32, LDK);
#pragma unroll
      for (int t = 0; t < 4; ++t)
        out[t] = wmma_bf16f32(pa,
                    load_frag_b(&smVt[(16 * t) * LDK + s2 * 32], LDK), out[t]);
    }
    __syncthreads();
  }

  // normalize + store to attn ws [B,S,1024] (= transpose(0,2,1,3).reshape)
#pragma unroll
  for (int t = 0; t < 4; ++t)
#pragma unroll
    for (int i = 0; i < 8; ++i) {
      const int r = qrow + hi * 8 + i;
      const int d = 16 * t + cl;
      attn[((size_t)b * S + r) * 1024 + h * 64 + d] = (bf16)(out[t][i] / lsum[i]);
    }
}

// ---------------------------------------------------------------------------
extern "C" void kernel_launch(void* const* d_in, const int* in_sizes, int n_in,
                              void* d_out, int out_size, void* d_ws, size_t ws_size,
                              hipStream_t stream) {
  const int B = 4, S = 2048, DIM = 1024, H = 16;
  const int M = B * S;                 // 8192
  const float* x     = (const float*)d_in[0];
  const float* w_in  = (const float*)d_in[1];
  const float* b_in  = (const float*)d_in[2];
  const float* w_out = (const float*)d_in[3];
  const float* b_out = (const float*)d_in[4];
  const unsigned char* kv_mask = (const unsigned char*)d_in[5];

  const size_t headElems = (size_t)B * H * S * 64;   // 8M elems = 16MB bf16
  bf16* q    = (bf16*)d_ws;
  bf16* k    = q + headElems;
  bf16* v    = k + headElems;
  bf16* attn = v + headElems;                        // [B,S,1024] bf16

  // 1) QKV projection: [8192,1024] x [1024,3072] -> scatter q/k/v + bias
  {
    dim3 g(M / 128, (3 * DIM) / 128);
    gemm_bf16_wmma<false, 0><<<g, 256, 0, stream>>>(
        x, w_in, b_in, nullptr, q, k, v, M, 3 * DIM, DIM, S);
  }
  // 2) RoPE on q,k in place
  {
    const size_t total = (size_t)2 * B * H * S * 32;
    rope_kernel<<<(unsigned)(total / 256), 256, 0, stream>>>(q, k, S);
  }
  // 3) Flash attention per (b, h, 128-q-row block)
  {
    dim3 g(S / 128, H, B);
    flash_attn<<<g, 256, 0, stream>>>(q, k, v, kv_mask, attn, S);
  }
  // 4) Output projection: [8192,1024] x [1024,1024] + bias -> f32 d_out
  {
    dim3 g(M / 128, DIM / 128);
    gemm_bf16_wmma<true, 1><<<g, 256, 0, stream>>>(
        attn, w_out, b_out, (float*)d_out, nullptr, nullptr, nullptr,
        M, DIM, DIM, S);
  }
}
